// GNN_mtl_gnn_5239860101893
// MI455X (gfx1250) — matrix operand order repaired and checked by reading, not compiled, not run on hardware
//
#include <hip/hip_runtime.h>
#include <hip/hip_bf16.h>

typedef __attribute__((ext_vector_type(16))) __bf16 v16bf;
typedef __attribute__((ext_vector_type(8)))  float  v8f;

#define WPB 8  // waves per block (256 threads, wave32)

// ---------------- weight conversion: f32 [dout][din] -> bf16 padded [doutpad][dinpad]
__global__ void cvt_w_kernel(const float* __restrict__ W, __bf16* __restrict__ Wb,
                             int dout, int din, int dinpad, int total) {
  int i = blockIdx.x * blockDim.x + threadIdx.x;
  if (i >= total) return;
  int o = i / dinpad;
  int k = i - o * dinpad;
  float v = (o < dout && k < din) ? W[o * din + k] : 0.0f;
  Wb[i] = (__bf16)v;
}

// ---------------- degree helpers
__global__ void fill_kernel(float* __restrict__ p, float v, int n) {
  int i = blockIdx.x * blockDim.x + threadIdx.x;
  if (i < n) p[i] = v;
}
__global__ void deg_count_kernel(const int* __restrict__ dst, float* __restrict__ deg, int e) {
  int i = blockIdx.x * blockDim.x + threadIdx.x;
  if (i < e) atomicAdd(&deg[dst[i]], 1.0f);
}
__global__ void rsqrt_kernel(float* __restrict__ p, int n) {
  int i = blockIdx.x * blockDim.x + threadIdx.x;
  if (i < n) p[i] = rsqrtf(p[i]);
}

// ---------------- GCN aggregation (hidden = 128)
// self-loop: out[i][:] = t[i][:] * dinv[i]^2   (one thread per float4)
__global__ void self_init_kernel(const float* __restrict__ t, const float* __restrict__ dinv,
                                 float* __restrict__ out, int n) {
  int idx = blockIdx.x * blockDim.x + threadIdx.x;
  int node = idx >> 5;
  if (node >= n) return;
  int c = (idx & 31) * 4;
  float di = dinv[node];
  float s = di * di;
  float4 v = *(const float4*)(t + node * 128 + c);
  float4 o = make_float4(v.x * s, v.y * s, v.z * s, v.w * s);
  *(float4*)(out + node * 128 + c) = o;
}

// one wave per edge: lane handles 4 channels; float4 gather + 4 f32 atomics
__global__ void edge_agg_kernel(const int* __restrict__ src, const int* __restrict__ dst,
                                const float* __restrict__ t, const float* __restrict__ dinv,
                                float* __restrict__ out, int e) {
  int lane = threadIdx.x & 31;
  int wave = threadIdx.x >> 5;
  int eid = blockIdx.x * WPB + wave;
  if (eid >= e) return;
  int s = src[eid];
  int d = dst[eid];
  float c = dinv[s] * dinv[d];
  float4 v = *(const float4*)(t + s * 128 + lane * 4);
  float* op = out + d * 128 + lane * 4;
  atomicAdd(op + 0, v.x * c);
  atomicAdd(op + 1, v.y * c);
  atomicAdd(op + 2, v.z * c);
  atomicAdd(op + 3, v.w * c);
}

// h[i][:] = relu(h[i][:] + b[:]) , hidden=128, one thread per float4
__global__ void bias_relu_kernel(float* __restrict__ h, const float* __restrict__ b, int n) {
  int idx = blockIdx.x * blockDim.x + threadIdx.x;
  int node = idx >> 5;
  if (node >= n) return;
  int c = (idx & 31) * 4;
  float4 v = *(const float4*)(h + node * 128 + c);
  float4 bb = *(const float4*)(b + c);
  v.x = fmaxf(v.x + bb.x, 0.0f);
  v.y = fmaxf(v.y + bb.y, 0.0f);
  v.z = fmaxf(v.z + bb.z, 0.0f);
  v.w = fmaxf(v.w + bb.w, 0.0f);
  *(float4*)(h + node * 128 + c) = v;
}

// ---------------- WMMA bf16 GEMM:  out[n, DOUT] = act(in[n, DIN] @ W.T + b) [+ res]
// Wb is bf16, padded [DOUTPAD][DINPAD] (DINPAD % 32 == 0, DOUTPAD % 16 == 0).
// One wave computes a 16-row strip across all DOUTPAD columns; the C tiles are
// staged through a per-wave LDS strip so final stores are coalesced float4.
template<int DIN, int DINPAD, int DOUT, int DOUTPAD, bool RELU, bool BIAS, bool RES>
__global__ void gemm_wmma_kernel(const float* __restrict__ in,
                                 const __bf16* __restrict__ Wb,
                                 const float* __restrict__ bias,
                                 const float* __restrict__ res,
                                 float* __restrict__ out, int nRows) {
  static_assert(DINPAD % 32 == 0 && DOUTPAD % 16 == 0 && DOUT % 4 == 0, "pad");
  constexpr int NCT = DOUTPAD / 16;

  __shared__ float smem[WPB * 16 * DOUTPAD];

  int lane = threadIdx.x & 31;
  int wave = threadIdx.x >> 5;
  int rowBase = (blockIdx.x * WPB + wave) * 16;
  if (rowBase >= nRows) return;

  v8f acc[NCT];
#pragma unroll
  for (int ct = 0; ct < NCT; ++ct)
#pragma unroll
    for (int r = 0; r < 8; ++r) acc[ct][r] = 0.0f;

  // A fragment addressing (contiguous K order, same as B):
  // lanes 0-15: row=lane,    elems 0..15 -> K 0..15
  // lanes 16-31: row=lane-16, elems 0..15 -> K 16..31
  int arow = rowBase + (lane & 15);
  bool arow_ok = (arow < nRows);
  int koffA = (lane < 16) ? 0 : 16;
  // B fragment addressing: lane = column (mod 16); lanes 0-15 K 0..15, lanes 16-31 K 16..31
  int colB = lane & 15;
  int koffB = (lane < 16) ? 0 : 16;

  for (int k0 = 0; k0 < DINPAD; k0 += 32) {
    v16bf a;
#pragma unroll
    for (int e = 0; e < 8; ++e) {
      int k1 = k0 + koffA + e;
      int k2 = k1 + 8;
      float f1 = (arow_ok && k1 < DIN) ? in[arow * DIN + k1] : 0.0f;
      float f2 = (arow_ok && k2 < DIN) ? in[arow * DIN + k2] : 0.0f;
      a[e]     = (__bf16)f1;
      a[e + 8] = (__bf16)f2;
    }
#pragma unroll
    for (int ct = 0; ct < NCT; ++ct) {
      v16bf bf = *(const v16bf*)(Wb + (ct * 16 + colB) * DINPAD + k0 + koffB);
      acc[ct] = __builtin_amdgcn_wmma_f32_16x16x32_bf16(
          false, a, false, bf, (short)0, acc[ct], false, false);
    }
  }

  // ---- epilogue: scatter C tiles (bias+relu applied) into the wave's LDS strip
  // C/D layout: elem r -> local row r + (lane<16?0:8), col = ct*16 + (lane&15)
  float* sm = smem + wave * 16 * DOUTPAD;
  int rhalf = (lane < 16) ? 0 : 8;
#pragma unroll
  for (int ct = 0; ct < NCT; ++ct) {
    int col = ct * 16 + colB;
    float bv = 0.0f;
    if (BIAS) bv = (DOUT == DOUTPAD || col < DOUT) ? bias[col] : 0.0f;
#pragma unroll
    for (int r = 0; r < 8; ++r) {
      float v = acc[ct][r] + bv;
      if (RELU) v = fmaxf(v, 0.0f);
      sm[(r + rhalf) * DOUTPAD + col] = v;
    }
  }
  // cross-lane LDS RAW within this wave: wait for LDS writes to land.
  // (cannot use __syncthreads(): tail waves may have exited early)
  asm volatile("s_wait_dscnt 0x0" ::: "memory");

  // ---- coalesced float4 stores (entire 512B rows per iteration)
  constexpr int NC4 = DOUT / 4;        // float4s per row
  constexpr int TOT4 = 16 * NC4;       // float4s per 16-row strip
  int nrem = nRows - rowBase;          // rows valid in this strip (>=1)
  for (int j = lane; j < TOT4; j += 32) {
    int row = j / NC4;
    int c4 = (j - row * NC4) * 4;
    if (row < nrem) {
      float4 v = *(const float4*)(sm + row * DOUTPAD + c4);
      if (RES) {
        float4 rv = *(const float4*)(res + (rowBase + row) * DOUT + c4);
        v.x += rv.x; v.y += rv.y; v.z += rv.z; v.w += rv.w;
      }
      *(float4*)(out + (rowBase + row) * DOUT + c4) = v;
    }
  }
}

static inline size_t align_up(size_t x, size_t a) { return (x + a - 1) & ~(a - 1); }

extern "C" void kernel_launch(void* const* d_in, const int* in_sizes, int n_in,
                              void* d_out, int out_size, void* d_ws, size_t ws_size,
                              hipStream_t stream) {
  const float* x   = (const float*)d_in[0];
  const int*   ei  = (const int*)d_in[1];   // [2, E] row-major: row0 = src, row1 = dst
  const float* W1  = (const float*)d_in[2];
  const float* b1  = (const float*)d_in[3];
  const float* W2  = (const float*)d_in[4];
  const float* b2  = (const float*)d_in[5];
  const float* W3  = (const float*)d_in[6];
  const float* b3  = (const float*)d_in[7];
  const float* W4  = (const float*)d_in[8];
  const float* b4  = (const float*)d_in[9];
  const float* Wc1 = (const float*)d_in[10];
  const float* bc1 = (const float*)d_in[11];
  const float* Wc2 = (const float*)d_in[12];
  const float* bc2 = (const float*)d_in[13];
  const float* W5  = (const float*)d_in[14];
  const float* b5  = (const float*)d_in[15];
  float* out = (float*)d_out;

  const int n = in_sizes[0] / 5;        // 100000
  const int E = in_sizes[1] / 2;        // 3200000
  const int* e_src = ei;
  const int* e_dst = ei + E;

  // workspace carve-out
  char* ws = (char*)d_ws;
  size_t off = 0;
  float* hA = (float*)(ws + off); off = align_up(off + (size_t)n * 128 * 4, 256);
  float* hB = (float*)(ws + off); off = align_up(off + (size_t)n * 128 * 4, 256);
  float* deg = (float*)(ws + off); off = align_up(off + (size_t)n * 4, 256);
  __bf16* W1b  = (__bf16*)(ws + off); off = align_up(off + 64 * 32 * 2, 256);
  __bf16* W2b  = (__bf16*)(ws + off); off = align_up(off + 128 * 64 * 2, 256);
  __bf16* W3b  = (__bf16*)(ws + off); off = align_up(off + 128 * 128 * 2, 256);
  __bf16* W4b  = (__bf16*)(ws + off); off = align_up(off + 128 * 128 * 2, 256);
  __bf16* Wc1b = (__bf16*)(ws + off); off = align_up(off + 128 * 128 * 2, 256);
  __bf16* Wc2b = (__bf16*)(ws + off); off = align_up(off + 128 * 128 * 2, 256);
  __bf16* W5b  = (__bf16*)(ws + off); off = align_up(off + 64 * 128 * 2, 256);
  (void)ws_size; (void)n_in; (void)out_size;

  // ---- weight conversion (tiny)
  cvt_w_kernel<<<(64 * 32 + 255) / 256, 256, 0, stream>>>(W1, W1b, 64, 5, 32, 64 * 32);
  cvt_w_kernel<<<(128 * 64 + 255) / 256, 256, 0, stream>>>(W2, W2b, 128, 64, 64, 128 * 64);
  cvt_w_kernel<<<(128 * 128 + 255) / 256, 256, 0, stream>>>(W3, W3b, 128, 128, 128, 128 * 128);
  cvt_w_kernel<<<(128 * 128 + 255) / 256, 256, 0, stream>>>(W4, W4b, 128, 128, 128, 128 * 128);
  cvt_w_kernel<<<(128 * 128 + 255) / 256, 256, 0, stream>>>(Wc1, Wc1b, 128, 128, 128, 128 * 128);
  cvt_w_kernel<<<(128 * 128 + 255) / 256, 256, 0, stream>>>(Wc2, Wc2b, 128, 128, 128, 128 * 128);
  cvt_w_kernel<<<(64 * 128 + 255) / 256, 256, 0, stream>>>(W5, W5b, 60, 128, 128, 64 * 128);

  // ---- degrees: deg = 1 (self loop) + indeg; then dinv = rsqrt(deg)
  fill_kernel<<<(n + 255) / 256, 256, 0, stream>>>(deg, 1.0f, n);
  deg_count_kernel<<<(E + 255) / 256, 256, 0, stream>>>(e_dst, deg, E);
  rsqrt_kernel<<<(n + 255) / 256, 256, 0, stream>>>(deg, n);

  const int rowTiles = (n + 15) / 16;
  const int gBlocks = (rowTiles + WPB - 1) / WPB;

  // L1: x[n,5] -> hB (as [n,64]),  relu
  gemm_wmma_kernel<5, 32, 64, 64, true, true, false>
      <<<gBlocks, 256, 0, stream>>>(x, W1b, b1, nullptr, hB, n);
  // L2: hB[n,64] -> hA[n,128], relu
  gemm_wmma_kernel<64, 64, 128, 128, true, true, false>
      <<<gBlocks, 256, 0, stream>>>(hB, W2b, b2, nullptr, hA, n);
  // L3: hB = relu(hA@W3.T+b3) + hA
  gemm_wmma_kernel<128, 128, 128, 128, true, true, true>
      <<<gBlocks, 256, 0, stream>>>(hA, W3b, b3, hA, hB, n);
  // L4: hA = relu(hB@W4.T+b4) + hB
  gemm_wmma_kernel<128, 128, 128, 128, true, true, true>
      <<<gBlocks, 256, 0, stream>>>(hB, W4b, b4, hB, hA, n);

  // ---- GCN conv 1: t = hA@Wc1.T (hB); aggregate into hA; +bias, relu
  gemm_wmma_kernel<128, 128, 128, 128, false, false, false>
      <<<gBlocks, 256, 0, stream>>>(hA, Wc1b, nullptr, nullptr, hB, n);
  self_init_kernel<<<((size_t)n * 32 + 255) / 256, 256, 0, stream>>>(hB, deg, hA, n);
  edge_agg_kernel<<<(E + WPB - 1) / WPB, 256, 0, stream>>>(e_src, e_dst, hB, deg, hA, E);
  bias_relu_kernel<<<((size_t)n * 32 + 255) / 256, 256, 0, stream>>>(hA, bc1, n);

  // ---- GCN conv 2
  gemm_wmma_kernel<128, 128, 128, 128, false, false, false>
      <<<gBlocks, 256, 0, stream>>>(hA, Wc2b, nullptr, nullptr, hB, n);
  self_init_kernel<<<((size_t)n * 32 + 255) / 256, 256, 0, stream>>>(hB, deg, hA, n);
  edge_agg_kernel<<<(E + WPB - 1) / WPB, 256, 0, stream>>>(e_src, e_dst, hB, deg, hA, E);
  bias_relu_kernel<<<((size_t)n * 32 + 255) / 256, 256, 0, stream>>>(hA, bc2, n);

  // ---- final: out[n,60] = hA @ W5.T + b5
  gemm_wmma_kernel<128, 128, 60, 64, false, true, false>
      <<<gBlocks, 256, 0, stream>>>(hA, W5b, b5, nullptr, out, n);
}
